// Lstm_82789789597912
// MI455X (gfx1250) — matrix operand order repaired and checked by reading, not compiled, not run on hardware
//
#include <hip/hip_runtime.h>
#include <cstdint>
#include <cstddef>

// ---------------------------------------------------------------------------
// LSTM (TF BasicLSTMCell order i,j,f,o; forget_bias=1.0) on gfx1250 via WMMA.
// B=128, T=128, EMB=HID=300, NOUT=3.
// Strategy: pack everything into V_WMMA_F32_16X16X32_F16 fragment layouts,
// one fused GEMM+gate kernel launch per timestep (launch boundary = device
// sync for the recurrence), f32 accumulation + f32 c/h state, f16 operands.
// ---------------------------------------------------------------------------

typedef __attribute__((ext_vector_type(16))) _Float16 v16h;
typedef __attribute__((ext_vector_type(8)))  float    v8f;

#define EMB   300
#define HID   300
#define NOUT  3
#define BATCH 128
#define TLEN  128

#define KCH 10   // K chunks of 32 per operand part (x or h): 300 padded to 320
#define MT  8    // batch tiles of 16 (128/16)
#define CT  19   // per-gate output-col tiles of 16 (300 padded to 304)

// A-fragment (16-bit, 16x32) K index for pair p (0..7) in a given lane-half.
// ISA 05_wmma §7.12.2: VGPR0..3 hold K={2p,2p+1}+8*half, VGPR4..7 hold +16.
__device__ __forceinline__ int a_pair_k(int p, int half) {
    int base = (p < 4) ? (2 * p) : (16 + 2 * (p - 4));
    return base + 8 * half;
}

// ---------------------------------------------------------------------------
// Pack W_lstm (600 x 1200 f32) into f16 B-fragments:
//   wpack[g][ct][kc(0..19)][lane][e], kc<10 = x-part rows, kc>=10 = h-part.
// B layout (16-bit, 32x16): lane half h, col n=lane&15, element e -> K = 8h+e (e<8), 16+8h+(e-8) (e>=8).
// ---------------------------------------------------------------------------
__global__ void pack_w_kernel(const float* __restrict__ W, _Float16* __restrict__ wpack) {
    int kc = blockIdx.x;            // 0..19
    int ct = blockIdx.y;            // 0..18
    int g  = blockIdx.z;            // gate 0..3 (i,j,f,o)
    int lane = threadIdx.x >> 4;    // 0..31
    int e    = threadIdx.x & 15;    // 0..15
    int half = lane >> 4;
    int n    = lane & 15;
    int K    = (e < 8) ? (half * 8 + e) : (16 + half * 8 + (e - 8)); // K within this 32-chunk
    int kpart = (kc >= KCH);                  // 0 = x rows, 1 = h rows
    int krow  = (kc - kpart * KCH) * 32 + K;  // row within the 300-part
    int col   = g * HID + ct * 16 + n;
    float v = 0.f;
    if ((ct * 16 + n) < HID && krow < 300) {
        v = W[(size_t)(kpart * 300 + krow) * (4 * HID) + col];
    }
    size_t idx = ((((size_t)g * CT + ct) * (2 * KCH) + kc) * 32 + lane) * 16 + e;
    wpack[idx] = (_Float16)v;
}

// ---------------------------------------------------------------------------
// Embedding gather straight into A-fragment layout:
//   xpack[t][mt][kc][lane][e] = f16(emb[x[b][t]][kc*32 + K(e,half)]), b=16*mt+m
// ---------------------------------------------------------------------------
__global__ void pack_x_kernel(const int* __restrict__ x, const float* __restrict__ emb,
                              _Float16* __restrict__ xpack) {
    int kc = blockIdx.x;   // 0..9
    int mt = blockIdx.y;   // 0..7
    int t  = blockIdx.z;   // 0..127
    int lane = threadIdx.x;
    int half = lane >> 4;
    int m    = lane & 15;
    int b    = mt * 16 + m;
    int row  = x[b * TLEN + t];
    const float* erow = emb + (size_t)row * EMB;
    v16h frag;
#pragma unroll
    for (int p = 0; p < 8; ++p) {
        int K0 = kc * 32 + a_pair_k(p, half);
        float v0 = (K0     < EMB) ? erow[K0]     : 0.f;
        float v1 = (K0 + 1 < EMB) ? erow[K0 + 1] : 0.f;
        frag[2 * p]     = (_Float16)v0;
        frag[2 * p + 1] = (_Float16)v1;
    }
    size_t idx = ((((size_t)t * MT + mt) * KCH + kc) * 32 + lane) * 16;
    *(v16h*)(xpack + idx) = frag;
}

// Zero c/h state and the h A-fragment buffer (padding stays zero forever).
__global__ void zero_kernel(float* c_state, float* h_state, _Float16* hpack) {
    int i = blockIdx.x * blockDim.x + threadIdx.x;
    if (i < BATCH * HID) { c_state[i] = 0.f; h_state[i] = 0.f; }
    if (i < MT * KCH * 32 * 16) hpack[i] = (_Float16)0.f;
}

// ---------------------------------------------------------------------------
// One LSTM timestep: z = [x_t | h] @ W + b, fused gate update.
// grid (CT, MT), block 128 = 4 waves; wave g computes gate g's 16x16 z-tile
// with 20 v_wmma_f32_16x16x32_f16 ops (K = 2*320), tiles meet in LDS, then
// 128 threads apply sigmoid/tanh + length mask and write c/h (+ h fragments).
// ---------------------------------------------------------------------------
__global__ void __launch_bounds__(128)
lstm_step_kernel(const _Float16* __restrict__ xpack,
                 const _Float16* __restrict__ wpack,
                 _Float16* __restrict__ hpack,
                 float* __restrict__ c_state,
                 float* __restrict__ h_state,
                 const float* __restrict__ b_lstm,
                 const int* __restrict__ lengths,
                 int t) {
    __shared__ float zl[4][16][16];
    int ct = blockIdx.x;
    int mt = blockIdx.y;
    int tid  = threadIdx.x;
    int g    = tid >> 5;    // gate / wave id
    int lane = tid & 31;

    const _Float16* xf = xpack + (((size_t)t * MT + mt) * KCH) * 32 * 16;
    const _Float16* hf = hpack + ((size_t)mt * KCH) * 32 * 16;
    const _Float16* wf = wpack + (((size_t)g * CT + ct) * (2 * KCH)) * 32 * 16;

    v8f acc = {};
#pragma unroll
    for (int kc = 0; kc < KCH; ++kc) {   // x-part, K = 0..319
        v16h a  = *(const v16h*)(xf + ((size_t)kc * 32 + lane) * 16);
        v16h bm = *(const v16h*)(wf + ((size_t)kc * 32 + lane) * 16);
        acc = __builtin_amdgcn_wmma_f32_16x16x32_f16(false, a, false, bm,
                                                     (short)0, acc, false, false);
    }
#pragma unroll
    for (int kc = 0; kc < KCH; ++kc) {   // h-part, K = 0..319
        v16h a  = *(const v16h*)(hf + ((size_t)kc * 32 + lane) * 16);
        v16h bm = *(const v16h*)(wf + ((size_t)(KCH + kc) * 32 + lane) * 16);
        acc = __builtin_amdgcn_wmma_f32_16x16x32_f16(false, a, false, bm,
                                                     (short)0, acc, false, false);
    }

    // C/D layout: lane half h, n = lane&15; VGPR v -> row m = v + 8h.
    int half = lane >> 4;
    int n    = lane & 15;
#pragma unroll
    for (int v = 0; v < 8; ++v) zl[g][v + 8 * half][n] = acc[v];
    __syncthreads();

    // Gate combine: 256 tile elements, 2 per thread.
#pragma unroll
    for (int r = 0; r < 2; ++r) {
        int idx = tid * 2 + r;
        int m  = idx >> 4;
        int nn = idx & 15;
        int col = ct * 16 + nn;
        if (col < HID) {
            int b = mt * 16 + m;
            float zi = zl[0][m][nn] + b_lstm[col];
            float zj = zl[1][m][nn] + b_lstm[HID + col];
            float zf = zl[2][m][nn] + b_lstm[2 * HID + col];
            float zo = zl[3][m][nn] + b_lstm[3 * HID + col];
            if (t < lengths[b]) {      // frozen past sequence length
                float c_old = c_state[b * HID + col];
                float si = 1.f / (1.f + __expf(-zi));
                float sf = 1.f / (1.f + __expf(-(zf + 1.f)));  // forget_bias = 1
                float so = 1.f / (1.f + __expf(-zo));
                float c_new = sf * c_old + si * tanhf(zj);
                float h_new = so * tanhf(c_new);
                c_state[b * HID + col] = c_new;
                h_state[b * HID + col] = h_new;
                // Scatter h into next step's A-fragment slot (inverse of a_pair_k).
                int kc    = col >> 5;
                int Kin   = col & 31;
                int hhalf = (Kin >> 3) & 1;
                int Kb    = Kin - 8 * hhalf;
                int p     = (Kb < 8) ? (Kb >> 1) : (4 + ((Kb - 16) >> 1));
                int e     = 2 * p + (Kin & 1);
                int hlane = hhalf * 16 + m;
                hpack[(((size_t)mt * KCH + kc) * 32 + hlane) * 16 + e] = (_Float16)h_new;
            }
        }
    }
}

// Final dense head: logits[b][j] = h[b,:] @ W_dense[:,j] + b_dense[j]
__global__ void dense_kernel(const float* __restrict__ h_state,
                             const float* __restrict__ Wd,
                             const float* __restrict__ bd,
                             float* __restrict__ out) {
    int tid = blockIdx.x * blockDim.x + threadIdx.x;
    if (tid >= BATCH * NOUT) return;
    int b = tid / NOUT, j = tid % NOUT;
    float s = bd[j];
    for (int k = 0; k < HID; ++k) s += h_state[b * HID + k] * Wd[k * NOUT + j];
    out[tid] = s;
}

extern "C" void kernel_launch(void* const* d_in, const int* in_sizes, int n_in,
                              void* d_out, int out_size, void* d_ws, size_t ws_size,
                              hipStream_t stream) {
    (void)in_sizes; (void)n_in; (void)out_size; (void)ws_size;
    const int*   x       = (const int*)d_in[0];
    const int*   lengths = (const int*)d_in[1];
    const float* emb     = (const float*)d_in[2];
    const float* W_lstm  = (const float*)d_in[3];
    const float* b_lstm  = (const float*)d_in[4];
    const float* W_dense = (const float*)d_in[5];
    const float* b_dense = (const float*)d_in[6];
    float* out = (float*)d_out;

    // Workspace layout (~12.4 MB total)
    char* ws = (char*)d_ws;
    auto align256 = [](size_t v) { return (v + 255) & ~(size_t)255; };
    size_t off = 0;
    _Float16* wpack = (_Float16*)(ws + off);
    off = align256(off + (size_t)4 * CT * (2 * KCH) * 32 * 16 * 2);   // 1.56 MB
    _Float16* xpack = (_Float16*)(ws + off);
    off = align256(off + (size_t)TLEN * MT * KCH * 32 * 16 * 2);      // 10 MB
    _Float16* hpack = (_Float16*)(ws + off);
    off = align256(off + (size_t)MT * KCH * 32 * 16 * 2);             // 80 KB
    float* c_state = (float*)(ws + off);
    off = align256(off + (size_t)BATCH * HID * 4);                    // 150 KB
    float* h_state = (float*)(ws + off);
    off = align256(off + (size_t)BATCH * HID * 4);                    // 150 KB

    zero_kernel<<<dim3((MT * KCH * 32 * 16 + 255) / 256), 256, 0, stream>>>(
        c_state, h_state, hpack);
    pack_w_kernel<<<dim3(2 * KCH, CT, 4), 512, 0, stream>>>(W_lstm, wpack);
    pack_x_kernel<<<dim3(KCH, MT, TLEN), 32, 0, stream>>>(x, emb, xpack);

    for (int t = 0; t < TLEN; ++t) {
        lstm_step_kernel<<<dim3(CT, MT), 128, 0, stream>>>(
            xpack, wpack, hpack, c_state, h_state, b_lstm, lengths, t);
    }
    dense_kernel<<<dim3((BATCH * NOUT + 127) / 128), 128, 0, stream>>>(
        h_state, W_dense, b_dense, out);
}
